// DomainAdaptationLayer_65833258713533
// MI455X (gfx1250) — compile-verified, matrix operand order checked by reading, not run on hardware
//
#include <hip/hip_runtime.h>
#include <hip/hip_bf16.h>

// ---------------- Problem constants ----------------
#define BB 4096
#define FF 512
#define JJ 16
#define MM 64
#define UU 512
#define JM (JJ * MM)   // 1024
// SIGMA = 0.5 -> 1/(2*sigma^2) = 2.0
#define INV2SIG2 2.0f

// ---------------- WMMA types ----------------
typedef __attribute__((ext_vector_type(16))) __bf16 v16bf;
typedef __attribute__((ext_vector_type(8)))  float  v8f;

union FragBF16 {
    uint4 q[2];   // two 16-byte chunks = 16 bf16 values
    v16bf v;
};

// ---------------- bf16 helpers (bit-level, round-to-nearest-even) ----------------
__device__ __forceinline__ unsigned short f2bf(float f) {
    unsigned u = __float_as_uint(f);
    unsigned r = u + 0x7FFFu + ((u >> 16) & 1u);
    return (unsigned short)(r >> 16);
}
__device__ __forceinline__ float bf2f(unsigned short s) {
    return __uint_as_float(((unsigned)s) << 16);
}

// Load one 16-bit WMMA A/B^T fragment per ISA 7.12.2 layout:
// lane l holds row (l&15); K-half (l>>4); two 16B chunks per 32-K step.
__device__ __forceinline__ void load_frag(FragBF16& f, const unsigned short* row,
                                          int k, int half) {
    f.q[0] = *(const uint4*)(row + k + half * 8);
    f.q[1] = *(const uint4*)(row + k + 16 + half * 8);
}

__device__ __forceinline__ v8f wmma_bf16(const FragBF16& a, const FragBF16& b, v8f c) {
    return __builtin_amdgcn_wmma_f32_16x16x32_bf16(
        false, a.v, false, b.v, (short)0, c, false, false);
}

// ---------------- Kernel 1a: h -> bf16, h2[b] = sum_f h^2 ----------------
__global__ void dal_conv_h(const float* __restrict__ h,
                           unsigned short* __restrict__ hbf,
                           float* __restrict__ h2) {
    const int row = blockIdx.x;
    const int tid = threadIdx.x;
    float s = 0.f;
    for (int i = tid; i < FF; i += 256) {
        float v = h[(size_t)row * FF + i];
        hbf[(size_t)row * FF + i] = f2bf(v);
        s = fmaf(v, v, s);
    }
    __shared__ float red[256];
    red[tid] = s;
    __syncthreads();
    for (int off = 128; off > 0; off >>= 1) {
        if (tid < off) red[tid] += red[tid + off];
        __syncthreads();
    }
    if (tid == 0) h2[row] = red[0];
}

// ---------------- Kernel 1b: domains -> bf16, d2[jm] = sum_f d^2 ----------------
__global__ void dal_conv_d(const float* __restrict__ dom,
                           unsigned short* __restrict__ dbf,
                           float* __restrict__ d2) {
    const int row = blockIdx.x;   // jm in [0,1024)
    const int tid = threadIdx.x;
    float s = 0.f;
    for (int i = tid; i < FF; i += 256) {
        float v = dom[(size_t)row * FF + i];
        dbf[(size_t)row * FF + i] = f2bf(v);
        s = fmaf(v, v, s);
    }
    __shared__ float red[256];
    red[tid] = s;
    __syncthreads();
    for (int off = 128; off > 0; off >>= 1) {
        if (tid < off) red[tid] += red[tid + off];
        __syncthreads();
    }
    if (tid == 0) d2[row] = red[0];
}

// ---------------- Kernel 1c: W[jm,u] -> Wt[u,jm] bf16 ----------------
__global__ void dal_transpose_w(const float* __restrict__ W,
                                unsigned short* __restrict__ wt) {
    const int jm = blockIdx.x;    // 1024 blocks
    for (int u = threadIdx.x; u < UU; u += 256) {
        wt[(size_t)u * JM + jm] = f2bf(W[(size_t)jm * UU + u]);
    }
}

// ---------------- Kernel 2: k_dd_mean[j] = mean_{m,n} exp(-2*||dm-dn||^2) ----------------
__global__ void dal_kdd(const float* __restrict__ dom,
                        const float* __restrict__ d2,
                        float* __restrict__ kdd) {
    const int j = blockIdx.x;     // 16 blocks
    const int tid = threadIdx.x;  // 256 threads
    float acc = 0.f;
    for (int p = tid; p < MM * MM; p += 256) {
        const int m = p >> 6, n = p & 63;
        const float* dm = dom + ((size_t)j * MM + m) * FF;
        const float* dn = dom + ((size_t)j * MM + n) * FF;
        float dot = 0.f;
        for (int f = 0; f < FF; ++f) dot = fmaf(dm[f], dn[f], dot);
        float sq = d2[j * MM + m] + d2[j * MM + n] - 2.f * dot;
        acc += __expf(-INV2SIG2 * sq);
    }
    __shared__ float red[256];
    red[tid] = acc;
    __syncthreads();
    for (int off = 128; off > 0; off >>= 1) {
        if (tid < off) red[tid] += red[tid + off];
        __syncthreads();
    }
    if (tid == 0) kdd[j] = red[0] * (1.f / (float)(MM * MM));
}

// ---------------- Kernel 3: GEMM1 (2x2 WMMA blocking) + Gaussian epilogue ----------------
// cross[b, jm] = h[b,:] . domains[jm,:], K_hd = exp(-2*(h2+d2-2*cross)) -> bf16
// One 32x32 output tile per wave, 8 waves/block. grid = (1024/32/8, 4096/32)
__global__ void dal_gemm1(const unsigned short* __restrict__ hbf,
                          const unsigned short* __restrict__ dbf,
                          const float* __restrict__ h2,
                          const float* __restrict__ d2,
                          unsigned short* __restrict__ khd) {
    const int lane  = threadIdx.x & 31;
    const int wave  = threadIdx.x >> 5;
    const int row0  = blockIdx.y * 32;
    const int col0  = (blockIdx.x * 8 + wave) * 32;
    const int half  = lane >> 4;
    const int idx16 = lane & 15;

    const unsigned short* arow0 = hbf + (size_t)(row0 + idx16) * FF;
    const unsigned short* arow1 = arow0 + (size_t)16 * FF;
    const unsigned short* brow0 = dbf + (size_t)(col0 + idx16) * FF;
    const unsigned short* brow1 = brow0 + (size_t)16 * FF;

    v8f c00 = {}, c01 = {}, c10 = {}, c11 = {};
    for (int k = 0; k < FF; k += 32) {
        FragBF16 a0, a1, b0, b1;
        load_frag(a0, arow0, k, half);
        load_frag(a1, arow1, k, half);
        load_frag(b0, brow0, k, half);
        load_frag(b1, brow1, k, half);
        c00 = wmma_bf16(a0, b0, c00);
        c01 = wmma_bf16(a0, b1, c01);
        c10 = wmma_bf16(a1, b0, c10);
        c11 = wmma_bf16(a1, b1, c11);
    }

    const int   colA  = col0 + idx16;
    const int   colB  = colA + 16;
    const float dcolA = d2[colA];
    const float dcolB = d2[colB];
    #pragma unroll
    for (int v = 0; v < 8; ++v) {
        const int r0 = row0 + v + 8 * half;
        const int r1 = r0 + 16;
        const float h2r0 = h2[r0];
        const float h2r1 = h2[r1];
        khd[(size_t)r0 * JM + colA] = f2bf(__expf(-INV2SIG2 * (h2r0 + dcolA - 2.f * c00[v])));
        khd[(size_t)r0 * JM + colB] = f2bf(__expf(-INV2SIG2 * (h2r0 + dcolB - 2.f * c01[v])));
        khd[(size_t)r1 * JM + colA] = f2bf(__expf(-INV2SIG2 * (h2r1 + dcolA - 2.f * c10[v])));
        khd[(size_t)r1 * JM + colB] = f2bf(__expf(-INV2SIG2 * (h2r1 + dcolB - 2.f * c11[v])));
    }
}

// ---------------- Kernel 4: per-row MMD softmax, scale K_hd in place ----------------
__global__ void dal_softmax_scale(unsigned short* __restrict__ khd,
                                  const float* __restrict__ kdd,
                                  float* __restrict__ prob) {
    const int b   = blockIdx.x;
    const int tid = threadIdx.x;            // 256 threads, 4 bf16 each (one uint2)
    __shared__ float jsum[JJ];
    __shared__ float jprob[JJ];
    if (tid < JJ) jsum[tid] = 0.f;
    __syncthreads();

    const int base = tid * 4;               // all 4 in same j (4 | 64)
    uint2* rowp = (uint2*)(khd + (size_t)b * JM + base);
    uint2 packed = *rowp;
    float vals[4];
    vals[0] = bf2f((unsigned short)(packed.x & 0xFFFFu));
    vals[1] = bf2f((unsigned short)(packed.x >> 16));
    vals[2] = bf2f((unsigned short)(packed.y & 0xFFFFu));
    vals[3] = bf2f((unsigned short)(packed.y >> 16));
    atomicAdd(&jsum[base >> 6], vals[0] + vals[1] + vals[2] + vals[3]);
    __syncthreads();

    if (tid == 0) {
        float logits[JJ];
        float mx = -1e30f;
        for (int j = 0; j < JJ; ++j) {
            float mmd = 1.f - 2.f * (jsum[j] * (1.f / (float)MM)) + kdd[j];
            logits[j] = -mmd;               // SOFTNESS = 1.0
            mx = fmaxf(mx, logits[j]);
        }
        float se = 0.f;
        for (int j = 0; j < JJ; ++j) { float e = __expf(logits[j] - mx); jprob[j] = e; se += e; }
        float inv = 1.f / se;
        for (int j = 0; j < JJ; ++j) { jprob[j] *= inv; prob[(size_t)b * JJ + j] = jprob[j]; }
    }
    __syncthreads();

    const float pj = jprob[base >> 6];
    unsigned lo = ((unsigned)f2bf(vals[1] * pj) << 16) | f2bf(vals[0] * pj);
    unsigned hi = ((unsigned)f2bf(vals[3] * pj) << 16) | f2bf(vals[2] * pj);
    *rowp = make_uint2(lo, hi);
}

// ---------------- Kernel 5: GEMM2 (2x2 WMMA blocking) + bias epilogue ----------------
// out[b,u] = (prob*K_hd)[b,:] . Wflat[:,u] + sum_j prob[b,j]*bias[j,u]
// grid = (512/32/8, 4096/32)
__global__ void dal_gemm2(const unsigned short* __restrict__ khd,
                          const unsigned short* __restrict__ wt,   // [U, JM] = B^T rows
                          const float* __restrict__ prob,
                          const float* __restrict__ bias,          // [J, U]
                          float* __restrict__ out) {
    const int lane  = threadIdx.x & 31;
    const int wave  = threadIdx.x >> 5;
    const int row0  = blockIdx.y * 32;
    const int col0  = (blockIdx.x * 8 + wave) * 32;
    const int half  = lane >> 4;
    const int idx16 = lane & 15;

    const unsigned short* arow0 = khd + (size_t)(row0 + idx16) * JM;
    const unsigned short* arow1 = arow0 + (size_t)16 * JM;
    const unsigned short* brow0 = wt + (size_t)(col0 + idx16) * JM;
    const unsigned short* brow1 = brow0 + (size_t)16 * JM;

    v8f c00 = {}, c01 = {}, c10 = {}, c11 = {};
    for (int k = 0; k < JM; k += 32) {
        FragBF16 a0, a1, b0, b1;
        load_frag(a0, arow0, k, half);
        load_frag(a1, arow1, k, half);
        load_frag(b0, brow0, k, half);
        load_frag(b1, brow1, k, half);
        c00 = wmma_bf16(a0, b0, c00);
        c01 = wmma_bf16(a0, b1, c01);
        c10 = wmma_bf16(a1, b0, c10);
        c11 = wmma_bf16(a1, b1, c11);
    }

    const int colA = col0 + idx16;
    const int colB = colA + 16;
    float blA[JJ], blB[JJ];
    #pragma unroll
    for (int j = 0; j < JJ; ++j) {
        blA[j] = bias[(size_t)j * UU + colA];
        blB[j] = bias[(size_t)j * UU + colB];
    }

    #pragma unroll
    for (int v = 0; v < 8; ++v) {
        const int r0 = row0 + v + 8 * half;
        const int r1 = r0 + 16;
        float p0[JJ], p1[JJ];
        const float4* pr0 = (const float4*)(prob + (size_t)r0 * JJ);
        const float4* pr1 = (const float4*)(prob + (size_t)r1 * JJ);
        #pragma unroll
        for (int q = 0; q < 4; ++q) {
            float4 t0 = pr0[q];
            p0[4 * q + 0] = t0.x; p0[4 * q + 1] = t0.y; p0[4 * q + 2] = t0.z; p0[4 * q + 3] = t0.w;
            float4 t1 = pr1[q];
            p1[4 * q + 0] = t1.x; p1[4 * q + 1] = t1.y; p1[4 * q + 2] = t1.z; p1[4 * q + 3] = t1.w;
        }
        float a00 = c00[v], a01 = c01[v], a10 = c10[v], a11 = c11[v];
        #pragma unroll
        for (int j = 0; j < JJ; ++j) {
            a00 = fmaf(p0[j], blA[j], a00);
            a01 = fmaf(p0[j], blB[j], a01);
            a10 = fmaf(p1[j], blA[j], a10);
            a11 = fmaf(p1[j], blB[j], a11);
        }
        out[(size_t)r0 * UU + colA] = a00;
        out[(size_t)r0 * UU + colB] = a01;
        out[(size_t)r1 * UU + colA] = a10;
        out[(size_t)r1 * UU + colB] = a11;
    }
}

// ---------------- Host launcher ----------------
extern "C" void kernel_launch(void* const* d_in, const int* in_sizes, int n_in,
                              void* d_out, int out_size, void* d_ws, size_t ws_size,
                              hipStream_t stream) {
    (void)in_sizes; (void)n_in; (void)out_size; (void)ws_size;
    const float* h    = (const float*)d_in[0];  // [4096, 512]
    const float* dom  = (const float*)d_in[1];  // [16, 64, 512]
    const float* W    = (const float*)d_in[2];  // [16, 64, 512] (flat [JM, U])
    const float* bias = (const float*)d_in[3];  // [16, 512]
    float* out = (float*)d_out;                 // [4096, 512]

    char* ws = (char*)d_ws;
    // workspace layout (all 64B aligned)
    unsigned short* hbf  = (unsigned short*)(ws + 0);                        // 4 MB
    unsigned short* dbf  = (unsigned short*)(ws + 4194304);                  // 1 MB
    unsigned short* wt   = (unsigned short*)(ws + 5242880);                  // 1 MB
    unsigned short* khd  = (unsigned short*)(ws + 6291456);                  // 8 MB
    float*          h2   = (float*)(ws + 14680064);                          // 16 KB
    float*          d2   = (float*)(ws + 14696448);                          // 4 KB
    float*          kdd  = (float*)(ws + 14700544);                          // 64 B
    float*          prob = (float*)(ws + 14700608);                          // 256 KB

    dal_conv_h<<<BB, 256, 0, stream>>>(h, hbf, h2);
    dal_conv_d<<<JM, 256, 0, stream>>>(dom, dbf, d2);
    dal_transpose_w<<<JM, 256, 0, stream>>>(W, wt);
    dal_kdd<<<JJ, 256, 0, stream>>>(dom, d2, kdd);
    dal_gemm1<<<dim3(JM / 32 / 8, BB / 32), 256, 0, stream>>>(hbf, dbf, h2, d2, khd);
    dal_softmax_scale<<<BB, 256, 0, stream>>>(khd, kdd, prob);
    dal_gemm2<<<dim3(UU / 32 / 8, BB / 32), 256, 0, stream>>>(khd, wt, prob, bias, out);
}
